// MultiHeadAttention_80255758893281
// MI455X (gfx1250) — compile-verified, matrix operand order checked by reading, not run on hardware
//
#include <hip/hip_runtime.h>

// MHA with cross-(head*batch) softmax (faithful to reference's implicit dim=0 softmax).
// bf16 WMMA (v_wmma_f32_16x16x32_bf16) for all GEMMs, fp32 accumulation.
// Pure-bf16 tile staging uses GLOBAL_LOAD_ASYNC_TO_LDS_B128 (ASYNCcnt path).

namespace {

constexpr int Sdim = 2048;
constexpr int Ddim = 1024;
constexpr int Hn   = 16;
constexpr int DKn  = 64;

typedef __attribute__((ext_vector_type(16))) __bf16 v16bf;
typedef __attribute__((ext_vector_type(8)))  float  v8f;

struct __align__(8)  U2 { unsigned int x, y; };
struct __align__(16) F4 { float x, y, z, w; };

__device__ __forceinline__ unsigned short f2bf(float f) {
  union { float f; unsigned int u; } c; c.f = f;
  unsigned int u = c.u;
  u += 0x7FFFu + ((u >> 16) & 1u);   // round-to-nearest-even
  return (unsigned short)(u >> 16);
}

__device__ __forceinline__ v8f wmma_bf16(v16bf a, v16bf b, v8f c) {
  // 8 args: (neg_a, A, neg_b, B, c_mod, C, reuse_a, reuse_b)
  return __builtin_amdgcn_wmma_f32_16x16x32_bf16(false, a, false, b, (short)0, c, false, false);
}

__device__ __forceinline__ v8f v8f_zero() {
  v8f z = {0.f, 0.f, 0.f, 0.f, 0.f, 0.f, 0.f, 0.f};
  return z;
}

// Async copy of 16 bytes global -> LDS (per lane), tracked by ASYNCcnt.
__device__ __forceinline__ void async_ld_b128(void* lds_dst, const void* gsrc) {
  unsigned lds_off = (unsigned)(unsigned long long)lds_dst;  // low 32b of LDS flat addr = LDS offset
  asm volatile("global_load_async_to_lds_b128 %0, %1, off"
               :: "v"(lds_off), "v"(gsrc) : "memory");
}
__device__ __forceinline__ void wait_async0() {
  asm volatile("s_wait_asynccnt 0x0" ::: "memory");
}

// Load a 16x32 bf16 fragment from an LDS tile (row-major, leading dim LD ushorts).
// Per ISA 7.12.2 16-bit A layout: lane L holds row (L&15); lanes 0-15 carry
// K={0..7,16..23}, lanes 16-31 carry K={8..15,24..31}. B operand uses the same
// pattern when its tile is stored N-major (i.e. as B^T).
template <int LD>
__device__ __forceinline__ v16bf load_frag(const unsigned short* lds, int row0, int col0) {
  const int lane = (int)(threadIdx.x & 31u);
  const int r  = row0 + (lane & 15);
  const int k0 = col0 + ((lane >> 4) << 3);
  union { v16bf v; unsigned int u[8]; } f;
  const unsigned int* p0 = (const unsigned int*)(lds + r * LD + k0);
  const unsigned int* p1 = (const unsigned int*)(lds + r * LD + k0 + 16);
  f.u[0] = p0[0]; f.u[1] = p0[1]; f.u[2] = p0[2]; f.u[3] = p0[3];
  f.u[4] = p1[0]; f.u[5] = p1[1]; f.u[6] = p1[2]; f.u[7] = p1[3];
  return f.v;
}

// ---------------------------------------------------------------------------
// Kernel 1: QKV projection.  C[m,n] = X[m,:] @ W[:,n] + bias[n]
// X: fp32 [4096,1024], W: fp32 [1024,1024] ([in,out]), out: bf16 [b][h][s][dk]
// grid (8, 32, 3), block 256
// ---------------------------------------------------------------------------
__global__ __launch_bounds__(256) void k_proj(
    const float* __restrict__ Xq, const float* __restrict__ Xk, const float* __restrict__ Xv,
    const float* __restrict__ Wq, const float* __restrict__ bq,
    const float* __restrict__ Wk, const float* __restrict__ bk,
    const float* __restrict__ Wv, const float* __restrict__ bv,
    unsigned short* __restrict__ Qb, unsigned short* __restrict__ Kb,
    unsigned short* __restrict__ Vb) {
  const float* X; const float* W; const float* bias; unsigned short* dst;
  if (blockIdx.z == 0)      { X = Xq; W = Wq; bias = bq; dst = Qb; }
  else if (blockIdx.z == 1) { X = Xk; W = Wk; bias = bk; dst = Kb; }
  else                      { X = Xv; W = Wv; bias = bv; dst = Vb; }

  __shared__ unsigned short As[128][40];  // [m][k]
  __shared__ unsigned short Bs[128][40];  // [n][k]  (W tile transposed)

  const int tid  = (int)threadIdx.x;
  const int lane = tid & 31;
  const int w    = tid >> 5;
  const int wm   = w & 3, wn = w >> 2;
  const int m0   = (int)blockIdx.y * 128;
  const int n0   = (int)blockIdx.x * 128;

  v8f acc[2][4];
  #pragma unroll
  for (int i = 0; i < 2; ++i)
    #pragma unroll
    for (int j = 0; j < 4; ++j) acc[i][j] = v8f_zero();

  for (int kt = 0; kt < Ddim; kt += 32) {
    #pragma unroll
    for (int rep = 0; rep < 4; ++rep) {          // X tile 128x32, fp32 -> bf16
      int idx = tid + rep * 256;
      int r = idx >> 3, c4 = (idx & 7) << 2;
      F4 x = *(const F4*)(X + (size_t)(m0 + r) * Ddim + kt + c4);
      As[r][c4 + 0] = f2bf(x.x); As[r][c4 + 1] = f2bf(x.y);
      As[r][c4 + 2] = f2bf(x.z); As[r][c4 + 3] = f2bf(x.w);
    }
    #pragma unroll
    for (int rep = 0; rep < 4; ++rep) {          // W tile 32x128 -> transposed
      int idx = tid + rep * 256;
      int kk = idx >> 5, n4 = (idx & 31) << 2;
      F4 ww = *(const F4*)(W + (size_t)(kt + kk) * Ddim + n0 + n4);
      Bs[n4 + 0][kk] = f2bf(ww.x); Bs[n4 + 1][kk] = f2bf(ww.y);
      Bs[n4 + 2][kk] = f2bf(ww.z); Bs[n4 + 3][kk] = f2bf(ww.w);
      if (kt + 32 < Ddim)
        __builtin_prefetch(W + (size_t)(kt + 32 + kk) * Ddim + n0 + n4, 0, 0);
    }
    __syncthreads();
    v16bf af[2], bf[4];
    #pragma unroll
    for (int i = 0; i < 2; ++i) af[i] = load_frag<40>(&As[0][0], wm * 32 + i * 16, 0);
    #pragma unroll
    for (int j = 0; j < 4; ++j) bf[j] = load_frag<40>(&Bs[0][0], wn * 64 + j * 16, 0);
    #pragma unroll
    for (int i = 0; i < 2; ++i)
      #pragma unroll
      for (int j = 0; j < 4; ++j) acc[i][j] = wmma_bf16(af[i], bf[j], acc[i][j]);
    __syncthreads();
  }

  const int hi8 = (lane >> 4) << 3;
  const int nl  = lane & 15;
  #pragma unroll
  for (int i = 0; i < 2; ++i)
    #pragma unroll
    for (int j = 0; j < 4; ++j) {
      int n  = n0 + wn * 64 + j * 16 + nl;
      float bb = bias[n];
      int h = n >> 6, dk = n & 63;
      #pragma unroll
      for (int r = 0; r < 8; ++r) {
        int m = m0 + wm * 32 + i * 16 + hi8 + r;
        int b = m >> 11, s = m & 2047;
        dst[(((size_t)(b * Hn + h)) * Sdim + s) * DKn + dk] = f2bf(acc[i][j][r] + bb);
      }
    }
}

// ---------------------------------------------------------------------------
// Kernel 2: denom[q,k] = sum over all 32 (h,b) of exp(Q_hb[q]·K_hb[k]/8)
// grid (16,16) (k-tile, q-tile), block 256
// ---------------------------------------------------------------------------
__global__ __launch_bounds__(256) void k_denom(const unsigned short* __restrict__ Qb,
                                               const unsigned short* __restrict__ Kb,
                                               float* __restrict__ denom) {
  __shared__ unsigned short Qs[128][72];  // [q][dk]  (144B pitch, 16B aligned)
  __shared__ unsigned short Ks[128][72];  // [k][dk]
  const int tid = (int)threadIdx.x, lane = tid & 31, w = tid >> 5;
  const int wm = w & 3, wn = w >> 2;
  const int q0 = (int)blockIdx.y * 128, k0 = (int)blockIdx.x * 128;

  float accD[2][4][8];
  #pragma unroll
  for (int i = 0; i < 2; ++i)
    #pragma unroll
    for (int j = 0; j < 4; ++j)
      #pragma unroll
      for (int r = 0; r < 8; ++r) accD[i][j][r] = 0.f;

  for (int hb = 0; hb < 32; ++hb) {
    const int h = hb >> 1, b = hb & 1;
    const unsigned short* qsrc = Qb + ((size_t)(b * Hn + h)) * Sdim * DKn;
    const unsigned short* ksrc = Kb + ((size_t)(b * Hn + h)) * Sdim * DKn;
    #pragma unroll
    for (int rep = 0; rep < 4; ++rep) {   // async bf16 copy: 128x64 = 1024 x b128
      int idx = tid + rep * 256;
      int r = idx >> 3, c8 = (idx & 7) << 3;
      async_ld_b128(&Qs[r][c8], qsrc + (size_t)(q0 + r) * DKn + c8);
      async_ld_b128(&Ks[r][c8], ksrc + (size_t)(k0 + r) * DKn + c8);
    }
    wait_async0();
    __syncthreads();

    v8f sacc[2][4];
    #pragma unroll
    for (int i = 0; i < 2; ++i)
      #pragma unroll
      for (int j = 0; j < 4; ++j) sacc[i][j] = v8f_zero();

    #pragma unroll
    for (int ks = 0; ks < 2; ++ks) {
      v16bf af[2], bf[4];
      #pragma unroll
      for (int i = 0; i < 2; ++i) af[i] = load_frag<72>(&Qs[0][0], wm * 32 + i * 16, ks * 32);
      #pragma unroll
      for (int j = 0; j < 4; ++j) bf[j] = load_frag<72>(&Ks[0][0], wn * 64 + j * 16, ks * 32);
      #pragma unroll
      for (int i = 0; i < 2; ++i)
        #pragma unroll
        for (int j = 0; j < 4; ++j) sacc[i][j] = wmma_bf16(af[i], bf[j], sacc[i][j]);
    }
    #pragma unroll
    for (int i = 0; i < 2; ++i)
      #pragma unroll
      for (int j = 0; j < 4; ++j)
        #pragma unroll
        for (int r = 0; r < 8; ++r)
          accD[i][j][r] += __expf(sacc[i][j][r] * 0.125f);
    __syncthreads();
  }

  const int hi8 = (lane >> 4) << 3, nl = lane & 15;
  #pragma unroll
  for (int i = 0; i < 2; ++i)
    #pragma unroll
    for (int j = 0; j < 4; ++j)
      #pragma unroll
      for (int r = 0; r < 8; ++r) {
        int q = q0 + wm * 32 + i * 16 + hi8 + r;
        int k = k0 + wn * 64 + j * 16 + nl;
        denom[(size_t)q * Sdim + k] = accD[i][j][r];
      }
}

// ---------------------------------------------------------------------------
// Kernel 3: ctx[b,q,h,:] = sum_k (exp(score)/denom[q,k]) * V_hb[k,:]
// grid (16, 32) = (q-tile, hb), block 256; streams keys in chunks of 64.
// ---------------------------------------------------------------------------
__global__ __launch_bounds__(256) void k_attnv(const unsigned short* __restrict__ Qb,
                                               const unsigned short* __restrict__ Kb,
                                               const unsigned short* __restrict__ Vb,
                                               const float* __restrict__ denom,
                                               unsigned short* __restrict__ ctx) {
  __shared__ unsigned short Qs[128][72];  // [q][dk]
  __shared__ unsigned short Ks[64][72];   // [k][dk]
  __shared__ unsigned short Vs[64][72];   // [dk][k]  (transposed => N-major)
  __shared__ unsigned short Ps[128][72];  // [q][k]   bf16 probabilities

  const int tid = (int)threadIdx.x, lane = tid & 31, w = tid >> 5;
  const int q0 = (int)blockIdx.x * 128;
  const int hb = (int)blockIdx.y;
  const int h = hb >> 1, b = hb & 1;
  const size_t base = ((size_t)(b * Hn + h)) * Sdim * DKn;
  const int hi8 = (lane >> 4) << 3, nl = lane & 15;

  #pragma unroll
  for (int rep = 0; rep < 4; ++rep) {     // stage Q tile once (async bf16 copy)
    int idx = tid + rep * 256;
    int r = idx >> 3, c8 = (idx & 7) << 3;
    async_ld_b128(&Qs[r][c8], Qb + base + (size_t)(q0 + r) * DKn + c8);
  }
  v8f oacc[4];
  #pragma unroll
  for (int j = 0; j < 4; ++j) oacc[j] = v8f_zero();
  wait_async0();
  __syncthreads();

  for (int kb = 0; kb < Sdim / 64; ++kb) {
    #pragma unroll
    for (int rep = 0; rep < 2; ++rep) {   // K chunk 64x64: async direct copy
      int idx = tid + rep * 256;
      int r = idx >> 3, c8 = (idx & 7) << 3;
      async_ld_b128(&Ks[r][c8], Kb + base + (size_t)(kb * 64 + r) * DKn + c8);
    }
    #pragma unroll
    for (int rep = 0; rep < 4; ++rep) {   // V chunk: manual transpose into LDS
      int idx = tid + rep * 256;
      int r = idx >> 4, c4 = (idx & 15) << 2;
      U2 d = *(const U2*)(Vb + base + (size_t)(kb * 64 + r) * DKn + c4);
      Vs[c4 + 0][r] = (unsigned short)(d.x & 0xffffu);
      Vs[c4 + 1][r] = (unsigned short)(d.x >> 16);
      Vs[c4 + 2][r] = (unsigned short)(d.y & 0xffffu);
      Vs[c4 + 3][r] = (unsigned short)(d.y >> 16);
    }
    wait_async0();
    __syncthreads();

    v8f sacc[4];                          // 16 q-rows x 64 keys per wave
    #pragma unroll
    for (int j = 0; j < 4; ++j) sacc[j] = v8f_zero();
    #pragma unroll
    for (int ks = 0; ks < 2; ++ks) {
      v16bf af = load_frag<72>(&Qs[0][0], w * 16, ks * 32);
      #pragma unroll
      for (int j = 0; j < 4; ++j) {
        v16bf bf = load_frag<72>(&Ks[0][0], j * 16, ks * 32);
        sacc[j] = wmma_bf16(af, bf, sacc[j]);
      }
    }
    #pragma unroll
    for (int j = 0; j < 4; ++j) {         // P = exp(s/8) / denom
      int kg = kb * 64 + j * 16 + nl;
      #pragma unroll
      for (int r = 0; r < 8; ++r) {
        int q = q0 + w * 16 + hi8 + r;
        float p = __expf(sacc[j][r] * 0.125f) / denom[(size_t)q * Sdim + kg];
        Ps[w * 16 + hi8 + r][j * 16 + nl] = f2bf(p);
      }
    }
    __syncthreads();

    #pragma unroll
    for (int ks = 0; ks < 2; ++ks) {      // O += P @ V
      v16bf af = load_frag<72>(&Ps[0][0], w * 16, ks * 32);
      #pragma unroll
      for (int j = 0; j < 4; ++j) {
        v16bf bf = load_frag<72>(&Vs[0][0], j * 16, ks * 32);
        oacc[j] = wmma_bf16(af, bf, oacc[j]);
      }
    }
    __syncthreads();
  }

  #pragma unroll
  for (int j = 0; j < 4; ++j)
    #pragma unroll
    for (int r = 0; r < 8; ++r) {
      int q = q0 + w * 16 + hi8 + r;
      int dk = j * 16 + nl;
      ctx[((size_t)(b * Sdim + q)) * Ddim + h * DKn + dk] = f2bf(oacc[j][r]);
    }
}

// ---------------------------------------------------------------------------
// Kernel 4: out = ctx(bf16) @ Wo + bo   -> fp32
// grid (8, 32), block 256
// ---------------------------------------------------------------------------
__global__ __launch_bounds__(256) void k_outproj(const unsigned short* __restrict__ Ctx,
                                                 const float* __restrict__ Wo,
                                                 const float* __restrict__ bo,
                                                 float* __restrict__ out) {
  __shared__ unsigned short As[128][40];  // 80B pitch, 16B aligned
  __shared__ unsigned short Bs[128][40];
  const int tid = (int)threadIdx.x, lane = tid & 31, w = tid >> 5;
  const int wm = w & 3, wn = w >> 2;
  const int m0 = (int)blockIdx.y * 128, n0 = (int)blockIdx.x * 128;

  v8f acc[2][4];
  #pragma unroll
  for (int i = 0; i < 2; ++i)
    #pragma unroll
    for (int j = 0; j < 4; ++j) acc[i][j] = v8f_zero();

  for (int kt = 0; kt < Ddim; kt += 32) {
    #pragma unroll
    for (int rep = 0; rep < 2; ++rep) {   // A tile 128x32 (bf16): async copy
      int idx = tid + rep * 256;
      int r = idx >> 2, c8 = (idx & 3) << 3;
      async_ld_b128(&As[r][c8], Ctx + (size_t)(m0 + r) * Ddim + kt + c8);
    }
    #pragma unroll
    for (int rep = 0; rep < 4; ++rep) {   // Wo tile, fp32 -> bf16, transposed
      int idx = tid + rep * 256;
      int kk = idx >> 5, n4 = (idx & 31) << 2;
      F4 ww = *(const F4*)(Wo + (size_t)(kt + kk) * Ddim + n0 + n4);
      Bs[n4 + 0][kk] = f2bf(ww.x); Bs[n4 + 1][kk] = f2bf(ww.y);
      Bs[n4 + 2][kk] = f2bf(ww.z); Bs[n4 + 3][kk] = f2bf(ww.w);
    }
    wait_async0();
    __syncthreads();
    v16bf af[2], bf[4];
    #pragma unroll
    for (int i = 0; i < 2; ++i) af[i] = load_frag<40>(&As[0][0], wm * 32 + i * 16, 0);
    #pragma unroll
    for (int j = 0; j < 4; ++j) bf[j] = load_frag<40>(&Bs[0][0], wn * 64 + j * 16, 0);
    #pragma unroll
    for (int i = 0; i < 2; ++i)
      #pragma unroll
      for (int j = 0; j < 4; ++j) acc[i][j] = wmma_bf16(af[i], bf[j], acc[i][j]);
    __syncthreads();
  }

  const int hi8 = (lane >> 4) << 3, nl = lane & 15;
  #pragma unroll
  for (int i = 0; i < 2; ++i)
    #pragma unroll
    for (int j = 0; j < 4; ++j) {
      int n = n0 + wn * 64 + j * 16 + nl;
      float bb = bo[n];
      #pragma unroll
      for (int r = 0; r < 8; ++r) {
        int m = m0 + wm * 32 + i * 16 + hi8 + r;
        out[(size_t)m * Ddim + n] = acc[i][j][r] + bb;
      }
    }
}

}  // namespace

extern "C" void kernel_launch(void* const* d_in, const int* in_sizes, int n_in,
                              void* d_out, int out_size, void* d_ws, size_t ws_size,
                              hipStream_t stream) {
  const float* query = (const float*)d_in[0];
  const float* key   = (const float*)d_in[1];
  const float* value = (const float*)d_in[2];
  const float* Wq = (const float*)d_in[3];
  const float* bq = (const float*)d_in[4];
  const float* Wk = (const float*)d_in[5];
  const float* bk = (const float*)d_in[6];
  const float* Wv = (const float*)d_in[7];
  const float* bv = (const float*)d_in[8];
  const float* Wo = (const float*)d_in[9];
  const float* bo = (const float*)d_in[10];

  const size_t QKV_ELEMS = (size_t)2 * Sdim * Ddim;  // B*S*D = 4,194,304
  unsigned short* Qb  = (unsigned short*)d_ws;
  unsigned short* Kb  = Qb + QKV_ELEMS;
  unsigned short* Vb  = Kb + QKV_ELEMS;
  unsigned short* ctx = Vb + QKV_ELEMS;
  float* denom = (float*)(ctx + QKV_ELEMS);          // [S][S] fp32, 16 MB

  dim3 blk(256);
  k_proj<<<dim3(Ddim / 128, (2 * Sdim) / 128, 3), blk, 0, stream>>>(
      query, key, value, Wq, bq, Wk, bk, Wv, bv, Qb, Kb, Vb);
  k_denom<<<dim3(Sdim / 128, Sdim / 128), blk, 0, stream>>>(Qb, Kb, denom);
  k_attnv<<<dim3(Sdim / 128, 32), blk, 0, stream>>>(Qb, Kb, Vb, denom, ctx);
  k_outproj<<<dim3(Ddim / 128, (2 * Sdim) / 128), blk, 0, stream>>>(ctx, Wo, bo, (float*)d_out);
}
